// OuterProductMean_56813827392293
// MI455X (gfx1250) — matrix operand order repaired and checked by reading, not compiled, not run on hardware
//
#include <hip/hip_runtime.h>
#include <hip/hip_bf16.h>

#define CS 256
#define CH 32
#define SD 128
#define RD 256
#define CZ 128
#define HK (CH * CH)   // 1024
#define JB 16          // j's per workgroup in stage 2

typedef __attribute__((ext_vector_type(16))) __bf16 v16bf;
typedef __attribute__((ext_vector_type(8)))  __bf16 v8bf;
typedef __attribute__((ext_vector_type(8)))  float  v8f;

static __device__ __forceinline__ v16bf join16(v8bf lo, v8bf hi) {
    v16bf r;
#pragma unroll
    for (int t = 0; t < 8; ++t) { r[t] = lo[t]; r[8 + t] = hi[t]; }
    return r;
}

// ---------------------------------------------------------------------------
// Kernel 1: LayerNorm over C_S + two projections to H, masked, bf16 output
// staged as [r][h][s] (s contiguous) for direct WMMA tile loads.
// One wave (32 lanes) per (s,r) row; 8 waves / block.
// ---------------------------------------------------------------------------
__global__ __launch_bounds__(256) void opm_ln_proj(
    const float* __restrict__ m,    const float* __restrict__ mask,
    const float* __restrict__ ln_w, const float* __restrict__ ln_b,
    const float* __restrict__ w1,   const float* __restrict__ b1,
    const float* __restrict__ w2,   const float* __restrict__ b2,
    __hip_bfloat16* __restrict__ Abuf, __hip_bfloat16* __restrict__ Bbuf)
{
    __shared__ float smh[8][CS];
    const int wave = threadIdx.x >> 5;
    const int lane = threadIdx.x & 31;
    const int row  = blockIdx.x * 8 + wave;   // row = s*RD + r
    const int s    = row >> 8;
    const int r    = row & 255;

    const float* mrow = m + (size_t)row * CS;
    float4 v0 = *reinterpret_cast<const float4*>(mrow + lane * 8);
    float4 v1 = *reinterpret_cast<const float4*>(mrow + lane * 8 + 4);
    float vals[8] = {v0.x, v0.y, v0.z, v0.w, v1.x, v1.y, v1.z, v1.w};

    float sum = 0.f, sq = 0.f;
#pragma unroll
    for (int k = 0; k < 8; ++k) { sum += vals[k]; sq += vals[k] * vals[k]; }
#pragma unroll
    for (int off = 16; off > 0; off >>= 1) {   // wave32 reduction
        sum += __shfl_xor(sum, off, 32);
        sq  += __shfl_xor(sq,  off, 32);
    }
    const float mu  = sum * (1.0f / CS);
    const float var = sq * (1.0f / CS) - mu * mu;
    const float rs  = rsqrtf(var + 1e-5f);

    const int cbase = lane * 8;
#pragma unroll
    for (int k = 0; k < 8; ++k) {
        const int c = cbase + k;
        smh[wave][c] = (vals[k] - mu) * rs * ln_w[c] + ln_b[c];
    }
    __syncthreads();

    const int h = lane;
    float acc1 = b1[h], acc2 = b2[h];
    const float* mh = smh[wave];
#pragma unroll 4
    for (int c = 0; c < CS; ++c) {
        const float v = mh[c];               // LDS broadcast
        acc1 += v * w1[c * CH + h];          // coalesced over lanes
        acc2 += v * w2[c * CH + h];
    }
    const float mk = mask[s * RD + r];
    const size_t oidx = ((size_t)r * CH + h) * SD + s;
    Abuf[oidx] = __float2bfloat16(acc1 * mk);
    Bbuf[oidx] = __float2bfloat16(acc2 * mk);
}

// ---------------------------------------------------------------------------
// Kernel 2: transpose+convert w_out [hk][z] f32 -> w_outT [z][hk] bf16 so the
// epilogue GEMM's B tiles (column z, K=hk contiguous) are straight 32B loads.
// ---------------------------------------------------------------------------
__global__ __launch_bounds__(256) void opm_wout_t(
    const float* __restrict__ w_out, __hip_bfloat16* __restrict__ w_outT)
{
    const int idx = blockIdx.x * 256 + threadIdx.x;   // 131072 elems
    const int hk  = idx >> 7;
    const int z   = idx & 127;
    w_outT[(size_t)z * HK + hk] = __float2bfloat16(w_out[idx]);
}

// ---------------------------------------------------------------------------
// Kernel 3: one workgroup per (i, j-block of 16).
//  Phase 1: outer tiles via WMMA:  M=32(h), N=512(16 pairs x 32 k), K=128(s);
//           64 wmma/wave; f32 accum -> bf16 LDS tile [pair][hk] (32 KB).
//  Phase 2: epilogue GEMM via WMMA: M=16(pairs), N=128(z), K=1024(hk);
//           64 wmma/wave; + mask-norm + bias + divide fused into the store.
// 4 waves / block; EXEC all-1s around every WMMA.
// ---------------------------------------------------------------------------
__global__ __launch_bounds__(128) void opm_outer(
    const __bf16* __restrict__ Abuf, const __bf16* __restrict__ Bbuf,
    const __bf16* __restrict__ w_outT,
    const float*  __restrict__ mask, const float* __restrict__ b_out,
    float* __restrict__ out)
{
    __shared__ __hip_bfloat16 sOut[JB * HK];  // 16 pairs x 1024 hk, bf16: 32 KB
    __shared__ float sNormPart[128];
    __shared__ float sNorm[JB];

    const int i    = blockIdx.x;
    const int jb   = blockIdx.y;
    const int tid  = threadIdx.x;
    const int wave = tid >> 5;
    const int lane = tid & 31;

    // ---- mask-norm partials: t -> (pair p = t>>3, s-chunk = t&7) ----
    {
        const int p  = tid >> 3;
        const int sc = tid & 7;
        const int jj = jb * JB + p;
        float part = 0.f;
#pragma unroll
        for (int s = sc * 16; s < sc * 16 + 16; ++s)
            part += mask[s * RD + i] * mask[s * RD + jj];
        sNormPart[tid] = part;
    }

    // ---- Phase 1: outer-product GEMM ----
    const __bf16* Ai = Abuf + (size_t)i * CH * SD;          // [h][s]
    const int am = lane & 15;              // A row within 16-row tile
    const int ak = (lane >> 4) * 16;       // A K sub-offset
    const int bn = lane & 15;              // B col within 16-col tile
    const int bk = (lane >> 4) * 16;       // B K sub-offset

    // Preload both 16x32 A tiles for all 4 K-steps (shared by every N-tile)
    v16bf avr[2][4];
#pragma unroll
    for (int mt = 0; mt < 2; ++mt)
#pragma unroll
        for (int ksi = 0; ksi < 4; ++ksi) {
            const __bf16* base = Ai + (mt * 16 + am) * SD + ksi * 32 + ak;
            avr[mt][ksi] = join16(*reinterpret_cast<const v8bf*>(base),
                                  *reinterpret_cast<const v8bf*>(base + 8));
        }

    for (int nt = 0; nt < 8; ++nt) {                // this wave's 8 N-tiles
        const int g  = wave * 8 + nt;               // global N-tile 0..31
        const int p  = g >> 1;                      // pair 0..15
        const int k0 = (g & 1) * 16;                // k offset within pair
        const __bf16* Bp = Bbuf + (size_t)(jb * JB + p) * CH * SD;  // [k][s]

        v8f acc0 = {}, acc1 = {};
#pragma unroll
        for (int ksi = 0; ksi < 4; ++ksi) {
            v16bf bv = *reinterpret_cast<const v16bf*>(
                Bp + (k0 + bn) * SD + ksi * 32 + bk);
            acc0 = __builtin_amdgcn_wmma_f32_16x16x32_bf16(
                       false, avr[0][ksi], false, bv, (short)0, acc0, false, false);
            acc1 = __builtin_amdgcn_wmma_f32_16x16x32_bf16(
                       false, avr[1][ksi], false, bv, (short)0, acc1, false, false);
        }
        // C layout: VGPR v -> M = v + (lane>=16 ? 8 : 0), N = lane%16
        const int kcol = k0 + (lane & 15);
        const int hrow = (lane >> 4) * 8;
#pragma unroll
        for (int v = 0; v < 8; ++v) {
            sOut[p * HK + (hrow + v) * CH + kcol]        = __float2bfloat16(acc0[v]);
            sOut[p * HK + (hrow + v + 16) * CH + kcol]   = __float2bfloat16(acc1[v]);
        }
    }
    __syncthreads();

    // reduce mask-norm partials (8 per pair)
    if (tid < JB) {
        float n = 0.f;
#pragma unroll
        for (int c = 0; c < 8; ++c) n += sNormPart[tid * 8 + c];
        sNorm[tid] = n;
    }
    __syncthreads();

    // ---- Phase 2: epilogue GEMM  M=16 pairs, N=128 z, K=1024 hk ----
    const __bf16* sA = reinterpret_cast<const __bf16*>(sOut);
    const int prow = lane & 15;            // pair = A row
    const int kha  = (lane >> 4) * 16;
    const int khb  = (lane >> 4) * 16;

    v8f eacc0 = {}, eacc1 = {};
    const int z0 = wave * 32;              // this wave's two z tiles
#pragma unroll 2
    for (int ksb = 0; ksb < HK; ksb += 32) {
        const __bf16* abase = sA + prow * HK + ksb + kha;
        v16bf av = join16(*reinterpret_cast<const v8bf*>(abase),
                          *reinterpret_cast<const v8bf*>(abase + 8));
        v16bf bv0 = *reinterpret_cast<const v16bf*>(
            w_outT + (size_t)(z0 + bn) * HK + ksb + khb);
        v16bf bv1 = *reinterpret_cast<const v16bf*>(
            w_outT + (size_t)(z0 + 16 + bn) * HK + ksb + khb);
        eacc0 = __builtin_amdgcn_wmma_f32_16x16x32_bf16(
                    false, av, false, bv0, (short)0, eacc0, false, false);
        eacc1 = __builtin_amdgcn_wmma_f32_16x16x32_bf16(
                    false, av, false, bv1, (short)0, eacc1, false, false);
    }

    // ---- fused bias / norm / store: M=pair, N=z ----
    const int pbase = (lane >> 4) * 8;
    const int zc    = lane & 15;
#pragma unroll
    for (int v = 0; v < 8; ++v) {
        const int p = pbase + v;
        const float inv = 1.0f / (1e-3f + sNorm[p]);
        const size_t rowo = ((size_t)i * RD + jb * JB + p) * CZ;
        const int za = z0 + zc;
        const int zb = z0 + 16 + zc;
        out[rowo + za] = (eacc0[v] + b_out[za]) * inv;
        out[rowo + zb] = (eacc1[v] + b_out[zb]) * inv;
    }
}

extern "C" void kernel_launch(void* const* d_in, const int* in_sizes, int n_in,
                              void* d_out, int out_size, void* d_ws, size_t ws_size,
                              hipStream_t stream) {
    const float* m     = (const float*)d_in[0];
    const float* mask  = (const float*)d_in[1];
    const float* ln_w  = (const float*)d_in[2];
    const float* ln_b  = (const float*)d_in[3];
    const float* w1    = (const float*)d_in[4];
    const float* b1    = (const float*)d_in[5];
    const float* w2    = (const float*)d_in[6];
    const float* b2    = (const float*)d_in[7];
    const float* w_out = (const float*)d_in[8];
    const float* b_out = (const float*)d_in[9];
    float* out = (float*)d_out;

    __hip_bfloat16* Abuf   = (__hip_bfloat16*)d_ws;              // 2 MB
    __hip_bfloat16* Bbuf   = Abuf + (size_t)RD * CH * SD;        // 2 MB
    __hip_bfloat16* w_outT = Bbuf + (size_t)RD * CH * SD;        // 256 KB

    // 32768 (s,r) rows, 8 waves/block
    opm_ln_proj<<<dim3(4096), dim3(256), 0, stream>>>(
        m, mask, ln_w, ln_b, w1, b1, w2, b2, Abuf, Bbuf);

    // w_out transpose -> bf16
    opm_wout_t<<<dim3(512), dim3(256), 0, stream>>>(w_out, w_outT);

    // one workgroup per (i, 16-wide j block)
    opm_outer<<<dim3(RD, RD / JB), dim3(128), 0, stream>>>(
        (const __bf16*)Abuf, (const __bf16*)Bbuf, (const __bf16*)w_outT,
        mask, b_out, out);
}
